// CriticNetwork_75995151336064
// MI455X (gfx1250) — compile-verified, matrix operand order checked by reading, not run on hardware
//
#include <hip/hip_runtime.h>

#define BATCH 8192

typedef __attribute__((ext_vector_type(16))) __bf16 v16bf;
typedef __attribute__((ext_vector_type(8)))  __bf16 v8bf;
typedef __attribute__((ext_vector_type(8)))  float  v8f;

// CDNA5 async global->LDS copies, tracked by ASYNCcnt (CDNA5 ISA).
__device__ __forceinline__ void async_copy_b128(void* lds, const void* gaddr) {
  unsigned l = (unsigned)(size_t)lds;
  unsigned long long g = (unsigned long long)(size_t)gaddr;
  asm volatile("global_load_async_to_lds_b128 %0, %1, off"
               :: "v"(l), "v"(g) : "memory");
}
__device__ __forceinline__ void async_copy_b64(void* lds, const void* gaddr) {
  unsigned l = (unsigned)(size_t)lds;
  unsigned long long g = (unsigned long long)(size_t)gaddr;
  asm volatile("global_load_async_to_lds_b64 %0, %1, off"
               :: "v"(l), "v"(g) : "memory");
}
// Wait until at most N async ops remain outstanding (in-order completion per wave).
template<int N>
__device__ __forceinline__ void wait_async_le() {
  asm volatile("s_wait_asynccnt %0" :: "n"(N) : "memory");
}

// ---------------- neighbor / adjacency-mask tables (from reference) ---------
__constant__ int NBRT[8][7] = {
  {1,2,3,4,5,6,7},{0,2,3,4,5,6,7},{0,1,3,4,5,6,7},{0,1,2,4,5,6,7},
  {0,1,2,3,5,6,7},{0,1,2,3,4,6,7},{0,1,2,3,4,5,7},{0,1,2,3,4,5,6}};
__constant__ int AROWT[8][7] = {
  {0,0,0,4,5,6,7},{0,2,3,4,5,6,7},{0,1,3,4,5,6,7},{0,1,2,4,5,6,7},
  {0,1,2,3,5,6,7},{0,1,2,3,4,6,7},{0,1,2,3,4,5,7},{0,1,2,3,4,5,6}};
__constant__ int ACOLT[8][7] = {
  {1,2,3,0,0,0,0},{1,1,1,1,1,1,1},{2,2,2,2,2,2,2},{3,3,3,3,3,3,3},
  {4,4,4,4,4,4,4},{5,5,5,5,5,5,5},{6,6,6,6,6,6,6},{7,7,7,7,7,7,7}};

// ---------------- weight transpose + f32->bf16 with K padding ---------------
// src: [batch][rows][cols] f32   dst: [batch][cols][rowsPad] bf16 (pad rows = 0)
__global__ void transpose_pad_bf16_kernel(const float* __restrict__ src,
                                          __bf16* __restrict__ dst,
                                          int batch, int rows, int cols, int rowsPad) {
  long idx = (long)blockIdx.x * blockDim.x + threadIdx.x;
  long total = (long)batch * cols * rowsPad;
  if (idx >= total) return;
  int bi  = (int)(idx / ((long)cols * rowsPad));
  int rem = (int)(idx - (long)bi * cols * rowsPad);
  int c  = rem / rowsPad;
  int rp = rem % rowsPad;
  float v = (rp < rows) ? src[((long)bi * rows + rp) * cols + c] : 0.f;
  dst[idx] = (__bf16)v;
}

// ---------------- pack Z1..Z8 node-major bf16 + head tail (576-wide) --------
__global__ void pack_inputs_kernel(const float* __restrict__ st,
                                   const float* __restrict__ ac,
                                   const float* __restrict__ z0, const float* __restrict__ z1,
                                   const float* __restrict__ z2, const float* __restrict__ z3,
                                   const float* __restrict__ z4, const float* __restrict__ z5,
                                   const float* __restrict__ z6, const float* __restrict__ z7,
                                   __bf16* __restrict__ Zcur,   // [8][B][3]
                                   __bf16* __restrict__ xbuf) { // [B][576]
  long b = (long)blockIdx.x * blockDim.x + threadIdx.x;
  if (b >= BATCH) return;
  const float* zs[8] = {z0,z1,z2,z3,z4,z5,z6,z7};
  #pragma unroll
  for (int n = 0; n < 8; ++n)
    #pragma unroll
    for (int k = 0; k < 3; ++k)
      Zcur[((long)n * BATCH + b) * 3 + k] = (__bf16)zs[n][b * 3 + k];
  #pragma unroll
  for (int j = 0; j < 24; ++j) xbuf[b * 576 + 512 + j] = (__bf16)st[b * 24 + j];
  #pragma unroll
  for (int j = 0; j < 16; ++j) xbuf[b * 576 + 536 + j] = (__bf16)ac[b * 16 + j];
  #pragma unroll
  for (int j = 552; j < 576; ++j) xbuf[b * 576 + j] = (__bf16)0.0f;
}

// ---------------- GAT attention + masked softmax + residual -----------------
// Z: [8][B][d] bf16 -> Zres: [8][B][dp] bf16 (dp>=d, pad zeroed)
__global__ __launch_bounds__(256) void gat_attention_kernel(
    const __bf16* __restrict__ Z, int d, int dp,
    const float* __restrict__ Wa,   // [8][7][7*d]
    const float* __restrict__ ba,   // [8][7]
    const float* __restrict__ Aadj, // [B][8][8]
    __bf16* __restrict__ Zres) {
  const int n = blockIdx.y;
  const long b = (long)blockIdx.x * blockDim.x + threadIdx.x;
  float logit[7];
  #pragma unroll
  for (int o = 0; o < 7; ++o) logit[o] = ba[n * 7 + o];
  const float* W = Wa + (long)n * 7 * 7 * d;
  const __bf16* zn[7];
  #pragma unroll
  for (int j = 0; j < 7; ++j)
    zn[j] = Z + ((long)NBRT[n][j] * BATCH + b) * d;
  for (int j = 0; j < 7; ++j) {
    const __bf16* zp = zn[j];
    const float* wj = W + (long)j * d;
    for (int k = 0; k < d; ++k) {
      float zv = (float)zp[k];
      #pragma unroll
      for (int o = 0; o < 7; ++o) logit[o] += zv * wj[(long)o * 7 * d + k];
    }
  }
  float mx = logit[0];
  #pragma unroll
  for (int o = 1; o < 7; ++o) mx = fmaxf(mx, logit[o]);
  float w[7], ssum = 0.f;
  #pragma unroll
  for (int o = 0; o < 7; ++o) { w[o] = __expf(logit[o] - mx); ssum += w[o]; }
  float inv = 1.f / ssum;
  #pragma unroll
  for (int o = 0; o < 7; ++o)
    w[o] *= inv * Aadj[b * 64 + AROWT[n][o] * 8 + ACOLT[n][o]];
  const __bf16* zs = Z    + ((long)n * BATCH + b) * d;
  __bf16*       zo = Zres + ((long)n * BATCH + b) * dp;
  for (int k = 0; k < d; ++k) {
    float acc = (float)zs[k];
    #pragma unroll
    for (int o = 0; o < 7; ++o) acc += w[o] * (float)zn[o][k];
    zo[k] = (__bf16)acc;
  }
  for (int k = d; k < dp; ++k) zo[k] = (__bf16)0.0f;
}

// ---------------- WMMA GEMM + bias + (relu/LN fused) ------------------------
// C[32 x H] per block; A: [node][M][K] bf16 (K % 32 == 0, zero padded),
// B: [node][H][K] bf16 (pre-transposed, zero padded).
// Double-buffered async global->LDS staging overlapped with WMMA compute.
// RELU_FIRST=true : out = LN(relu(A*W + b))   (GNN layers)
// RELU_FIRST=false: out = relu(LN(A*W + b))   (head layers)
template<int H, bool RELU_FIRST>
__global__ __launch_bounds__(256) void gemm_bias_act_ln(
    const __bf16* __restrict__ Aall, long aNodeStride, int K,
    const __bf16* __restrict__ Ball,
    const float* __restrict__ biasAll,
    const float* __restrict__ gammaAll,
    const float* __restrict__ betaAll,
    __bf16* __restrict__ Out, long outNodeStride, long outRowStride) {
  constexpr int BM  = 32;
  constexpr int NT  = H / 16;           // N tiles
  constexpr int TPW = (2 * NT) / 8;     // WMMA tiles per wave (8 waves)
  constexpr int NSTAGE = 1 + H / 64;    // async ops issued per wave per stage

  // staging buffers and LN tile are not live simultaneously -> overlap in LDS.
  // alignas(16) so v8bf fragment loads lower to ds_load_b128.
  union alignas(16) SH {
    struct { __bf16 la[2][BM][32]; __bf16 lb[2][H][32]; } s;
    float cbuf[BM][H];
  };
  __shared__ SH sh;

  const int tid  = threadIdx.x;
  const int wave = tid >> 5;
  const int lane = tid & 31;
  const int half = lane >> 4;
  const int r    = lane & 15;
  const int z    = blockIdx.z;
  const long rowBase = (long)blockIdx.x * BM;

  const __bf16* Ap = Aall + (long)z * aNodeStride;
  const __bf16* Bp = Ball + (long)z * H * K;
  const float* bias  = biasAll  + (long)z * H;
  const float* gamma = gammaAll + (long)z * H;
  const float* beta  = betaAll  + (long)z * H;

  const int mt = (wave * TPW) / NT;     // one M tile per wave
  int nt[TPW];
  #pragma unroll
  for (int i = 0; i < TPW; ++i) nt[i] = (wave * TPW + i) % NT;

  v8f acc[TPW];
  #pragma unroll
  for (int i = 0; i < TPW; ++i)
    #pragma unroll
    for (int j = 0; j < 8; ++j) acc[i][j] = 0.f;

  // stage one 32-wide K slab into buffer `buf` (every wave issues NSTAGE asyncs)
  auto stage = [&](int buf, int k0) {
    { // A tile: 32x32 bf16 = 256 x 8B chunks, exactly one per thread
      int rr = tid >> 3, cc = (tid & 7) * 4;
      async_copy_b64(&sh.s.la[buf][rr][cc], Ap + (rowBase + rr) * K + k0 + cc);
    }
    #pragma unroll
    for (int i = 0; i < H / 64; ++i) {  // B tile: Hx32 bf16 = H*4 x 16B chunks
      int c = tid + i * 256;
      int nn = c >> 2, cc = (c & 3) * 8;
      async_copy_b128(&sh.s.lb[buf][nn][cc], Bp + (long)nn * K + k0 + cc);
    }
  };

  const int ksteps = K / 32;
  stage(0, 0);
  for (int it = 0; it < ksteps; ++it) {
    const int cur = it & 1;
    if (it + 1 < ksteps) {
      stage(cur ^ 1, (it + 1) * 32);
      wait_async_le<NSTAGE>();          // previous slab (cur) complete for this wave
    } else {
      wait_async_le<0>();
    }
    __syncthreads();                    // cur slab visible to all waves

    union { v16bf v; v8bf h[2]; } af;
    af.h[0] = *(const v8bf*)__builtin_assume_aligned(&sh.s.la[cur][mt * 16 + r][0], 16);
    af.h[1] = *(const v8bf*)__builtin_assume_aligned(&sh.s.la[cur][mt * 16 + r][16], 16);
    {
      // select this lane's K-halves: lanes 0-15 K:{0..7,16..23}, lanes 16-31 K:{8..15,24..31}
      const v8bf* arow = (const v8bf*)__builtin_assume_aligned(&sh.s.la[cur][mt * 16 + r][0], 16);
      af.h[0] = arow[half];       // 16B chunk: K = half*8 .. half*8+7
      af.h[1] = arow[2 + half];   // 16B chunk: K = 16+half*8 .. 16+half*8+7
    }
    #pragma unroll
    for (int i = 0; i < TPW; ++i) {
      const v8bf* brow = (const v8bf*)__builtin_assume_aligned(&sh.s.lb[cur][nt[i] * 16 + r][0], 16);
      union { v16bf v; v8bf h[2]; } bfg;
      bfg.h[0] = brow[half];
      bfg.h[1] = brow[2 + half];
      acc[i] = __builtin_amdgcn_wmma_f32_16x16x32_bf16(
          false, af.v, false, bfg.v, (short)0, acc[i], false, false);
    }
    __syncthreads();                    // all waves done reading before overwrite
  }

  // D layout: VGPR j -> M = j + half*8, N = r
  #pragma unroll
  for (int i = 0; i < TPW; ++i) {
    int col = nt[i] * 16 + r;
    float bv = bias[col];
    #pragma unroll
    for (int j = 0; j < 8; ++j)
      sh.cbuf[mt * 16 + half * 8 + j][col] = acc[i][j] + bv;
  }
  __syncthreads();

  // fused LayerNorm: 8 lanes per row
  constexpr int CH = H / 8;
  const int row = tid >> 3;
  const int sub = tid & 7;
  float s = 0.f, s2 = 0.f;
  #pragma unroll
  for (int c = 0; c < CH; ++c) {
    float v = sh.cbuf[row][sub * CH + c];
    if (RELU_FIRST) v = fmaxf(v, 0.f);
    s += v; s2 += v * v;
  }
  #pragma unroll
  for (int m = 1; m < 8; m <<= 1) {
    s  += __shfl_xor(s,  m, 8);
    s2 += __shfl_xor(s2, m, 8);
  }
  const float mean = s / (float)H;
  const float var  = s2 / (float)H - mean * mean;
  const float rstd = rsqrtf(var + 1e-5f);

  __bf16* orow = Out + (long)z * outNodeStride + (rowBase + row) * outRowStride;
  #pragma unroll
  for (int c = 0; c < CH; ++c) {
    int col = sub * CH + c;
    float v = sh.cbuf[row][col];
    if (RELU_FIRST) v = fmaxf(v, 0.f);
    float y = (v - mean) * rstd * gamma[col] + beta[col];
    if (!RELU_FIRST) y = fmaxf(y, 0.f);
    orow[col] = (__bf16)y;
  }
}

// ---------------- final 64 -> 1 ---------------------------------------------
__global__ void head_final_kernel(const __bf16* __restrict__ x,
                                  const float* __restrict__ w,
                                  const float* __restrict__ b,
                                  float* __restrict__ out) {
  long i = (long)blockIdx.x * blockDim.x + threadIdx.x;
  if (i >= BATCH) return;
  float acc = b[0];
  #pragma unroll
  for (int k = 0; k < 64; ++k) acc += (float)x[i * 64 + k] * w[k];
  out[i] = acc;
}

// ---------------- launch ----------------------------------------------------
extern "C" void kernel_launch(void* const* d_in, const int* in_sizes, int n_in,
                              void* d_out, int out_size, void* d_ws, size_t ws_size,
                              hipStream_t stream) {
  (void)in_sizes; (void)n_in; (void)out_size; (void)ws_size;
  const float* state  = (const float*)d_in[0];
  const float* action = (const float*)d_in[1];
  const float* Zin[8];
  for (int i = 0; i < 8; ++i) Zin[i] = (const float*)d_in[2 + i];
  const float* Aadj = (const float*)d_in[10];
  // params in dict insertion order
  const float* Wa[3] = {(const float*)d_in[11], (const float*)d_in[17], (const float*)d_in[23]};
  const float* ba[3] = {(const float*)d_in[12], (const float*)d_in[18], (const float*)d_in[24]};
  const float* Wl[3] = {(const float*)d_in[13], (const float*)d_in[19], (const float*)d_in[25]};
  const float* bl[3] = {(const float*)d_in[14], (const float*)d_in[20], (const float*)d_in[26]};
  const float* g[3]  = {(const float*)d_in[15], (const float*)d_in[21], (const float*)d_in[27]};
  const float* be[3] = {(const float*)d_in[16], (const float*)d_in[22], (const float*)d_in[28]};
  const float* qw[4] = {(const float*)d_in[29], (const float*)d_in[33], (const float*)d_in[37], (const float*)d_in[41]};
  const float* qb[4] = {(const float*)d_in[30], (const float*)d_in[34], (const float*)d_in[38], (const float*)d_in[42]};
  const float* hg[3] = {(const float*)d_in[31], (const float*)d_in[35], (const float*)d_in[39]};
  const float* hb[3] = {(const float*)d_in[32], (const float*)d_in[36], (const float*)d_in[40]};

  // workspace carve
  char* basep = (char*)d_ws;
  size_t off = 0;
  auto carve = [&](size_t bytes) -> void* {
    void* p = basep + off;
    off += (bytes + 255) & ~(size_t)255;
    return p;
  };
  __bf16* zA   = (__bf16*)carve((size_t)8 * BATCH * 256 * 2); // activations [8][B][<=256]
  __bf16* zR   = (__bf16*)carve((size_t)8 * BATCH * 256 * 2); // attention out (padded K)
  __bf16* xbuf = (__bf16*)carve((size_t)BATCH * 576 * 2);     // head input, K padded 552->576
  __bf16* hb1  = (__bf16*)carve((size_t)BATCH * 256 * 2);
  __bf16* hb2  = (__bf16*)carve((size_t)BATCH * 128 * 2);
  __bf16* hb3  = (__bf16*)carve((size_t)BATCH * 64 * 2);
  __bf16* wT0  = (__bf16*)carve((size_t)8 * 256 * 32 * 2);    // [8][256][Kp=32]
  __bf16* wT1  = (__bf16*)carve((size_t)8 * 128 * 256 * 2);   // [8][128][256]
  __bf16* wT2  = (__bf16*)carve((size_t)8 * 64 * 128 * 2);    // [8][64][128]
  __bf16* qT0  = (__bf16*)carve((size_t)256 * 576 * 2);       // [256][Kp=576]
  __bf16* qT1  = (__bf16*)carve((size_t)128 * 256 * 2);
  __bf16* qT2  = (__bf16*)carve((size_t)64 * 128 * 2);

  auto launchT = [&](const float* src, __bf16* dst, int batch, int rows, int cols, int rowsPad) {
    long total = (long)batch * cols * rowsPad;
    transpose_pad_bf16_kernel<<<(unsigned)((total + 255) / 256), 256, 0, stream>>>(
        src, dst, batch, rows, cols, rowsPad);
  };
  launchT(Wl[0], wT0, 8, 3,   256, 32);
  launchT(Wl[1], wT1, 8, 256, 128, 256);
  launchT(Wl[2], wT2, 8, 128, 64,  128);
  launchT(qw[0], qT0, 1, 552, 256, 576);
  launchT(qw[1], qT1, 1, 256, 128, 256);
  launchT(qw[2], qT2, 1, 128, 64,  128);

  pack_inputs_kernel<<<BATCH / 256, 256, 0, stream>>>(
      state, action, Zin[0], Zin[1], Zin[2], Zin[3], Zin[4], Zin[5], Zin[6], Zin[7],
      zA, xbuf);

  dim3 attG(BATCH / 256, 8);
  dim3 gemmG8(BATCH / 32, 1, 8);
  dim3 gemmG1(BATCH / 32, 1, 1);

  // GNN layer 0 : d=3 (padded to 32) -> 256
  gat_attention_kernel<<<attG, 256, 0, stream>>>(zA, 3, 32, Wa[0], ba[0], Aadj, zR);
  gemm_bias_act_ln<256, true><<<gemmG8, 256, 0, stream>>>(
      zR, (long)BATCH * 32, 32, wT0, bl[0], g[0], be[0], zA, (long)BATCH * 256, 256);
  // GNN layer 1 : 256 -> 128
  gat_attention_kernel<<<attG, 256, 0, stream>>>(zA, 256, 256, Wa[1], ba[1], Aadj, zR);
  gemm_bias_act_ln<128, true><<<gemmG8, 256, 0, stream>>>(
      zR, (long)BATCH * 256, 256, wT1, bl[1], g[1], be[1], zA, (long)BATCH * 128, 128);
  // GNN layer 2 : 128 -> 64, scatter into xbuf columns [n*64 .. n*64+63] (stride 576)
  gat_attention_kernel<<<attG, 256, 0, stream>>>(zA, 128, 128, Wa[2], ba[2], Aadj, zR);
  gemm_bias_act_ln<64, true><<<gemmG8, 256, 0, stream>>>(
      zR, (long)BATCH * 128, 128, wT2, bl[2], g[2], be[2], xbuf, 64, 576);

  // head MLP
  gemm_bias_act_ln<256, false><<<gemmG1, 256, 0, stream>>>(
      xbuf, 0, 576, qT0, qb[0], hg[0], hb[0], hb1, 0, 256);
  gemm_bias_act_ln<128, false><<<gemmG1, 256, 0, stream>>>(
      hb1, 0, 256, qT1, qb[1], hg[1], hb[1], hb2, 0, 128);
  gemm_bias_act_ln<64, false><<<gemmG1, 256, 0, stream>>>(
      hb2, 0, 128, qT2, qb[2], hg[2], hb[2], hb3, 0, 64);

  head_final_kernel<<<BATCH / 256, 256, 0, stream>>>(hb3, qw[3], qb[3], (float*)d_out);
}
